// Block_29403346109123
// MI455X (gfx1250) — matrix operand order repaired and checked by reading, not compiled, not run on hardware
//
#include <hip/hip_runtime.h>

// ---------------------------------------------------------------------------
// Transformer encoder block for MI455X (gfx1250, wave32, WMMA).
// Compute-bound (~120 GFLOP vs ~150 MB HBM traffic): all GEMMs + attention
// einsums on v_wmma_f32_16x16x32_f16 (f32 accum). Weights pre-converted and
// pre-transposed to f16 once so tile staging is pure 16B copies, issued as
// gfx1250 async global->LDS loads (ASYNCcnt) when the builtin is available.
// ---------------------------------------------------------------------------

typedef __attribute__((ext_vector_type(8)))  _Float16 h8;
typedef __attribute__((ext_vector_type(16))) _Float16 v16h;
typedef __attribute__((ext_vector_type(8)))  float    v8f;
typedef int v4i __attribute__((vector_size(16)));

#define HDIM  512
#define NHEAD 8
#define HEADD 64
#define SEQ   4096
#define MTOT  8192      // B*S
#define FF    2048      // 4*H

#if defined(__has_builtin)
#if __has_builtin(__builtin_amdgcn_global_load_async_to_lds_b128) && \
    __has_builtin(__builtin_amdgcn_s_wait_asynccnt)
#define USE_ASYNC_LDS 1
#endif
#endif

__device__ __forceinline__ v8f wmma32(v16h a, v16h b, v8f c) {
  return __builtin_amdgcn_wmma_f32_16x16x32_f16(false, a, false, b, (short)0, c,
                                                false, false);
}
__device__ __forceinline__ v16h cat16(h8 lo, h8 hi) {
  return __builtin_shufflevector(lo, hi, 0,1,2,3,4,5,6,7,8,9,10,11,12,13,14,15);
}
__device__ __forceinline__ v8f vzero8() {
  v8f z = {0.f,0.f,0.f,0.f,0.f,0.f,0.f,0.f};
  return z;
}

// 16-byte global -> LDS copy; async (gfx1250 GLOBAL_LOAD_ASYNC_TO_LDS_B128)
// when the builtin exists, otherwise load+store fallback.
__device__ __forceinline__ void copy16(const _Float16* __restrict__ g, _Float16* l) {
#ifdef USE_ASYNC_LDS
  __builtin_amdgcn_global_load_async_to_lds_b128(
      (__attribute__((address_space(1))) v4i*)g,
      (__attribute__((address_space(3))) v4i*)l, 0, 0);
#else
  *(h8*)l = *(const h8*)g;
#endif
}
__device__ __forceinline__ void async_join() {
#ifdef USE_ASYNC_LDS
  __builtin_amdgcn_s_wait_asynccnt(0);
#endif
}

// ---------------------------------------------------------------------------
// Weight prep: W[K][N] fp32 -> WT[N][K] f16 (32x32 LDS tile transpose).
// ---------------------------------------------------------------------------
__global__ void wconv_kernel(const float* __restrict__ W, _Float16* __restrict__ WT,
                             int K, int N) {
  __shared__ _Float16 t[32][33];
  const int k0 = blockIdx.y * 32, n0 = blockIdx.x * 32;
  const int tid = threadIdx.x;
  for (int i = tid; i < 1024; i += 256) {
    const int r = i >> 5, c = i & 31;
    t[r][c] = (_Float16)W[(size_t)(k0 + r) * N + n0 + c];
  }
  __syncthreads();
  for (int i = tid; i < 1024; i += 256) {
    const int r = i >> 5, c = i & 31;       // r: n-local, c: k-local
    WT[(size_t)(n0 + r) * K + k0 + c] = t[c][r];
  }
}

// ---------------------------------------------------------------------------
// LayerNorm: one row (512) per 256-thread block, writes f16 output.
// ---------------------------------------------------------------------------
__global__ void ln_kernel(const float* __restrict__ x, const float* __restrict__ g,
                          const float* __restrict__ be, _Float16* __restrict__ out) {
  __shared__ float rsum[256];
  __shared__ float rsq[256];
  const int row = blockIdx.x;
  const int tid = threadIdx.x;
  const float* xr = x + (size_t)row * HDIM;
  float a = xr[tid];
  float b = xr[tid + 256];
  rsum[tid] = a + b;
  rsq[tid]  = a * a + b * b;
  __syncthreads();
  for (int s = 128; s > 0; s >>= 1) {
    if (tid < s) { rsum[tid] += rsum[tid + s]; rsq[tid] += rsq[tid + s]; }
    __syncthreads();
  }
  const float mu  = rsum[0] * (1.0f / (float)HDIM);
  const float var = rsq[0] * (1.0f / (float)HDIM) - mu * mu;
  const float rs  = rsqrtf(var + 1e-5f);
  _Float16* o = out + (size_t)row * HDIM;
  o[tid]       = (_Float16)((a - mu) * rs * g[tid]       + be[tid]);
  o[tid + 256] = (_Float16)((b - mu) * rs * g[tid + 256] + be[tid + 256]);
}

// ---------------------------------------------------------------------------
// WMMA GEMM core: 256 threads / 8 waves, 256x64 block tile, BK=32.
// A: f16 row-major [M,K]; BT: f16 [N][K] (pre-transposed weights).
// Wave w owns a 32x64 tile: 2 A-fragments x 4 B-fragments -> 8 WMMA / K-step.
// ---------------------------------------------------------------------------
__device__ __forceinline__ void gemm_core(const _Float16* __restrict__ A, int lda,
                                          const _Float16* __restrict__ BT, int ldb,
                                          int Ktot, int m0, int n0, v8f acc[2][4]) {
  __shared__ _Float16 sA[256][40];   // pad keeps every 8-half chunk 16B aligned
  __shared__ _Float16 sBT[64][40];
  const int tid  = threadIdx.x;
  const int lane = tid & 31;
  const int w    = tid >> 5;
  const int kh8  = (lane >> 4) * 16;   // A-fragment chunk base
  const int kb   = (lane >> 4) * 16;   // B-fragment chunk base
  const int nlo  = lane & 15;

  for (int k0 = 0; k0 < Ktot; k0 += 32) {
    // stage A tile: 256 rows x 32 halves = 1024 chunks of 8 halves
    for (int c = tid; c < 1024; c += 256) {
      const int row = c >> 2, c8 = c & 3;
      copy16(A + (size_t)(m0 + row) * lda + k0 + c8 * 8, &sA[row][c8 * 8]);
    }
    // stage B tile: 64 n-rows x 32 halves = 256 chunks (1 per thread)
    {
      const int row = tid >> 2, c8 = tid & 3;
      copy16(BT + (size_t)(n0 + row) * ldb + k0 + c8 * 8, &sBT[row][c8 * 8]);
    }
    async_join();
    __syncthreads();

    const v16h af0 = cat16(*(const h8*)&sA[w * 32 + nlo][kh8],
                           *(const h8*)&sA[w * 32 + nlo][kh8 + 8]);
    const v16h af1 = cat16(*(const h8*)&sA[w * 32 + 16 + nlo][kh8],
                           *(const h8*)&sA[w * 32 + 16 + nlo][kh8 + 8]);
#pragma unroll
    for (int nt = 0; nt < 4; ++nt) {
      const v16h bf = cat16(*(const h8*)&sBT[nt * 16 + nlo][kb],
                            *(const h8*)&sBT[nt * 16 + nlo][kb + 8]);
      acc[0][nt] = wmma32(af0, bf, acc[0][nt]);
      acc[1][nt] = wmma32(af1, bf, acc[1][nt]);
    }
    __syncthreads();
  }
}

#define GEMM_EPILOGUE_VARS                                   \
  const int lane = threadIdx.x & 31, w = threadIdx.x >> 5;   \
  (void)lane; (void)w;

// ---------------------------------------------------------------------------
// QKV projection: x_ln @ {Wq,Wk,Wv}^T-staged + bias.
// Q,K -> head-major f16 [B,NH,S,HD] (Q pre-scaled by 1/8);
// V   -> head-transposed f16 [B,NH,HD,S] (ready-made B^T for P*V).
// ---------------------------------------------------------------------------
__global__ void qkv_kernel(const _Float16* __restrict__ xln,
                           const _Float16* __restrict__ WTq, const float* __restrict__ bq,
                           const _Float16* __restrict__ WTk, const float* __restrict__ bk,
                           const _Float16* __restrict__ WTv, const float* __restrict__ bv,
                           _Float16* __restrict__ qh, _Float16* __restrict__ kh,
                           _Float16* __restrict__ vt) {
  const int z = blockIdx.z;
  const _Float16* WT = (z == 0) ? WTq : (z == 1) ? WTk : WTv;
  const float* bias  = (z == 0) ? bq  : (z == 1) ? bk  : bv;
  const float scale  = (z == 0) ? 0.125f : 1.0f;

  const int m0 = blockIdx.y * 256, n0 = blockIdx.x * 64;
  v8f acc[2][4];
#pragma unroll
  for (int mt = 0; mt < 2; ++mt)
#pragma unroll
    for (int nt = 0; nt < 4; ++nt) acc[mt][nt] = vzero8();
  gemm_core(xln, HDIM, WT, HDIM, HDIM, m0, n0, acc);

  GEMM_EPILOGUE_VARS
#pragma unroll
  for (int mt = 0; mt < 2; ++mt) {
#pragma unroll
    for (int nt = 0; nt < 4; ++nt) {
#pragma unroll
      for (int r = 0; r < 8; ++r) {
        const int row = m0 + w * 32 + mt * 16 + r + 8 * (lane >> 4);
        const int col = n0 + nt * 16 + (lane & 15);
        const float v = (acc[mt][nt][r] + bias[col]) * scale;
        const int b = row >> 12, s = row & (SEQ - 1);
        const int hh = col >> 6, hd = col & (HEADD - 1);
        const size_t bh = (size_t)(b * NHEAD + hh);
        if (z == 2) {
          vt[(bh * HEADD + hd) * SEQ + s] = (_Float16)v;
        } else {
          _Float16* dst = (z == 0) ? qh : kh;
          dst[(bh * SEQ + s) * HEADD + hd] = (_Float16)v;
        }
      }
    }
  }
}

// ---------------------------------------------------------------------------
// Flash attention: 128 threads / 4 waves per (b*NH, 64-query tile).
// Q fragments in registers for the whole kernel; 32-key K/V tiles in LDS.
// ---------------------------------------------------------------------------
__global__ void attn_kernel(const _Float16* __restrict__ qh,
                            const _Float16* __restrict__ kh,
                            const _Float16* __restrict__ vt,
                            _Float16* __restrict__ ctx) {
  __shared__ _Float16 sK[32][72];      // [key][hd] == B^T for Q*K^T
  __shared__ _Float16 sV[64][40];      // [hd][key] == B^T for P*V
  __shared__ _Float16 sP[4][16][40];   // per-wave P scratch (16 rows x 32 k)

  const int bh = blockIdx.y;           // b*NH + h
  const int b = bh >> 3, h = bh & 7;
  const int q0 = blockIdx.x * 64;
  const int tid = threadIdx.x, lane = tid & 31, w = tid >> 5;
  const size_t headoff = (size_t)bh * SEQ * HEADD;

  const int mrow = q0 + w * 16 + (lane & 15);
  const int kh8  = (lane >> 4) * 16;
  const int kb   = (lane >> 4) * 16;
  const int nlo  = lane & 15;
  const _Float16* qrow = qh + headoff + (size_t)mrow * HEADD;
  const v16h qf0 = cat16(*(const h8*)(qrow + kh8),      *(const h8*)(qrow + kh8 + 8));
  const v16h qf1 = cat16(*(const h8*)(qrow + 32 + kh8), *(const h8*)(qrow + 32 + kh8 + 8));

  v8f acc[4] = {vzero8(), vzero8(), vzero8(), vzero8()};
  float mrun[8], lrun[8];
#pragma unroll
  for (int r = 0; r < 8; ++r) { mrun[r] = -1e30f; lrun[r] = 0.0f; }

  for (int kt = 0; kt < SEQ; kt += 32) {
    // stage K tile (32 keys x 64 hd, row-major): 256 chunks of 8 halves
    for (int c = tid; c < 256; c += 128) {
      const int key = c >> 3, c8 = c & 7;
      copy16(kh + headoff + (size_t)(kt + key) * HEADD + c8 * 8, &sK[key][c8 * 8]);
    }
    // stage V tile from head-transposed layout: rows = hd, cols = 32 keys
    for (int c = tid; c < 256; c += 128) {
      const int hd = c >> 2, c8 = c & 3;
      copy16(vt + (headoff + (size_t)hd * SEQ) + kt + c8 * 8, &sV[hd][c8 * 8]);
    }
    async_join();
    __syncthreads();

    // logits for 32 keys: D0 = keys 0..15, D1 = keys 16..31 (K=64 over hd)
    v8f d0 = vzero8(), d1 = vzero8();
    {
      const v16h kf0 = cat16(*(const h8*)&sK[nlo][kb],      *(const h8*)&sK[nlo][kb + 8]);
      const v16h kf1 = cat16(*(const h8*)&sK[nlo][32 + kb], *(const h8*)&sK[nlo][32 + kb + 8]);
      d0 = wmma32(qf0, kf0, d0);
      d0 = wmma32(qf1, kf1, d0);
      const v16h kg0 = cat16(*(const h8*)&sK[16 + nlo][kb],      *(const h8*)&sK[16 + nlo][kb + 8]);
      const v16h kg1 = cat16(*(const h8*)&sK[16 + nlo][32 + kb], *(const h8*)&sK[16 + nlo][32 + kb + 8]);
      d1 = wmma32(qf0, kg0, d1);
      d1 = wmma32(qf1, kg1, d1);
    }

    // online softmax; row reductions across the 16-lane half-wave
#pragma unroll
    for (int r = 0; r < 8; ++r) {
      float t = fmaxf(d0[r], d1[r]);
      t = fmaxf(t, __shfl_xor(t, 1, 32));
      t = fmaxf(t, __shfl_xor(t, 2, 32));
      t = fmaxf(t, __shfl_xor(t, 4, 32));
      t = fmaxf(t, __shfl_xor(t, 8, 32));
      const float mn = fmaxf(mrun[r], t);
      const float al = __expf(mrun[r] - mn);
      const float p0 = __expf(d0[r] - mn);
      const float p1 = __expf(d1[r] - mn);
      float srow = p0 + p1;
      srow += __shfl_xor(srow, 1, 32);
      srow += __shfl_xor(srow, 2, 32);
      srow += __shfl_xor(srow, 4, 32);
      srow += __shfl_xor(srow, 8, 32);
      lrun[r] = lrun[r] * al + srow;
      mrun[r] = mn;
      const int prow = r + 8 * (lane >> 4);
      sP[w][prow][nlo]      = (_Float16)p0;
      sP[w][prow][16 + nlo] = (_Float16)p1;
#pragma unroll
      for (int nt = 0; nt < 4; ++nt) acc[nt][r] *= al;
    }

    // P (16x32) as A-fragment from wave-private LDS, then ctx += P*V
    const _Float16* pr = &sP[w][lane & 15][0];
    const v16h pf = cat16(*(const h8*)(pr + kh8), *(const h8*)(pr + kh8 + 8));
#pragma unroll
    for (int nt = 0; nt < 4; ++nt) {
      const v16h vf = cat16(*(const h8*)&sV[nt * 16 + nlo][kb],
                            *(const h8*)&sV[nt * 16 + nlo][kb + 8]);
      acc[nt] = wmma32(pf, vf, acc[nt]);
    }
    __syncthreads();
  }

  // normalize and store ctx in token-major f16 [M, H]
#pragma unroll
  for (int r = 0; r < 8; ++r) {
    const float inv = 1.0f / lrun[r];
    const int qrow2 = q0 + w * 16 + r + 8 * (lane >> 4);
    const size_t base = ((size_t)b * SEQ + qrow2) * HDIM + h * HEADD;
#pragma unroll
    for (int nt = 0; nt < 4; ++nt)
      ctx[base + nt * 16 + (lane & 15)] = (_Float16)(acc[nt][r] * inv);
  }
}

// ---------------------------------------------------------------------------
// Out-projection + residual: h = hidden + ctx @ Wo + bo   (f32 out)
// ---------------------------------------------------------------------------
__global__ void oproj_kernel(const _Float16* __restrict__ ctx,
                             const _Float16* __restrict__ WTo, const float* __restrict__ bo,
                             const float* __restrict__ hidden, float* __restrict__ hres) {
  const int m0 = blockIdx.y * 256, n0 = blockIdx.x * 64;
  v8f acc[2][4];
#pragma unroll
  for (int mt = 0; mt < 2; ++mt)
#pragma unroll
    for (int nt = 0; nt < 4; ++nt) acc[mt][nt] = vzero8();
  gemm_core(ctx, HDIM, WTo, HDIM, HDIM, m0, n0, acc);
  GEMM_EPILOGUE_VARS
#pragma unroll
  for (int mt = 0; mt < 2; ++mt)
#pragma unroll
    for (int nt = 0; nt < 4; ++nt)
#pragma unroll
      for (int r = 0; r < 8; ++r) {
        const int row = m0 + w * 32 + mt * 16 + r + 8 * (lane >> 4);
        const int col = n0 + nt * 16 + (lane & 15);
        const size_t o = (size_t)row * HDIM + col;
        hres[o] = hidden[o] + acc[mt][nt][r] + bo[col];
      }
}

// ---------------------------------------------------------------------------
// MLP-1: act = gelu_exact(y_ln @ W1 + b1)  -> f16 [M, 4H]
// ---------------------------------------------------------------------------
__global__ void mlp1_kernel(const _Float16* __restrict__ yln,
                            const _Float16* __restrict__ WT1, const float* __restrict__ b1,
                            _Float16* __restrict__ act) {
  const int m0 = blockIdx.y * 256, n0 = blockIdx.x * 64;
  v8f acc[2][4];
#pragma unroll
  for (int mt = 0; mt < 2; ++mt)
#pragma unroll
    for (int nt = 0; nt < 4; ++nt) acc[mt][nt] = vzero8();
  gemm_core(yln, HDIM, WT1, HDIM, HDIM, m0, n0, acc);
  GEMM_EPILOGUE_VARS
#pragma unroll
  for (int mt = 0; mt < 2; ++mt)
#pragma unroll
    for (int nt = 0; nt < 4; ++nt)
#pragma unroll
      for (int r = 0; r < 8; ++r) {
        const int row = m0 + w * 32 + mt * 16 + r + 8 * (lane >> 4);
        const int col = n0 + nt * 16 + (lane & 15);
        const float v = acc[mt][nt][r] + b1[col];
        const float ge = 0.5f * v * (1.0f + erff(v * 0.70710678118654752f));
        act[(size_t)row * FF + col] = (_Float16)ge;
      }
}

// ---------------------------------------------------------------------------
// MLP-2 + residual: out = h + act @ W2 + b2   (f32 out)
// ---------------------------------------------------------------------------
__global__ void mlp2_kernel(const _Float16* __restrict__ act,
                            const _Float16* __restrict__ WT2, const float* __restrict__ b2,
                            const float* __restrict__ hres, float* __restrict__ out) {
  const int m0 = blockIdx.y * 256, n0 = blockIdx.x * 64;
  v8f acc[2][4];
#pragma unroll
  for (int mt = 0; mt < 2; ++mt)
#pragma unroll
    for (int nt = 0; nt < 4; ++nt) acc[mt][nt] = vzero8();
  gemm_core(act, FF, WT2, FF, FF, m0, n0, acc);
  GEMM_EPILOGUE_VARS
#pragma unroll
  for (int mt = 0; mt < 2; ++mt)
#pragma unroll
    for (int nt = 0; nt < 4; ++nt)
#pragma unroll
      for (int r = 0; r < 8; ++r) {
        const int row = m0 + w * 32 + mt * 16 + r + 8 * (lane >> 4);
        const int col = n0 + nt * 16 + (lane & 15);
        const size_t o = (size_t)row * HDIM + col;
        out[o] = hres[o] + acc[mt][nt][r] + b2[col];
      }
}

// ---------------------------------------------------------------------------
// Workspace layout (bytes):
//   xln  f16 [8192,512]   @ 0
//   qh   f16 [B,NH,S,HD]  @ 8 MiB
//   kh   f16              @ 16 MiB
//   vt   f16 [B,NH,HD,S]  @ 24 MiB
//   ctx  f16 [8192,512]   @ 32 MiB
//   hres f32 [8192,512]   @ 40 MiB (16 MiB)
//   yln  f16 [8192,512]   @ 56 MiB
//   act  f16 [8192,2048]  @ 64 MiB (32 MiB)
//   WTq/WTk/WTv/WTo f16 [512,512]   @ 96/96.5/97/97.5 MiB
//   WT1 f16 [2048,512]    @ 98 MiB,  WT2 f16 [512,2048] @ 100 MiB  -> 102 MiB
// ---------------------------------------------------------------------------
extern "C" void kernel_launch(void* const* d_in, const int* in_sizes, int n_in,
                              void* d_out, int out_size, void* d_ws, size_t ws_size,
                              hipStream_t stream) {
  const float* hidden = (const float*)d_in[0];
  const float* Wq = (const float*)d_in[1];  const float* bq = (const float*)d_in[2];
  const float* Wk = (const float*)d_in[3];  const float* bk = (const float*)d_in[4];
  const float* Wv = (const float*)d_in[5];  const float* bv = (const float*)d_in[6];
  const float* Wo = (const float*)d_in[7];  const float* bo = (const float*)d_in[8];
  const float* W1 = (const float*)d_in[9];  const float* b1 = (const float*)d_in[10];
  const float* W2 = (const float*)d_in[11]; const float* b2 = (const float*)d_in[12];
  const float* g1 = (const float*)d_in[13]; const float* be1 = (const float*)d_in[14];
  const float* g2 = (const float*)d_in[15]; const float* be2 = (const float*)d_in[16];
  float* out = (float*)d_out;

  char* ws = (char*)d_ws;
  const size_t SZ16 = (size_t)MTOT * HDIM * sizeof(_Float16);   // 8 MiB
  const size_t SZW  = (size_t)HDIM * HDIM * sizeof(_Float16);   // 512 KiB
  _Float16* xln = (_Float16*)(ws);
  _Float16* qh  = (_Float16*)(ws + 1 * SZ16);
  _Float16* kh  = (_Float16*)(ws + 2 * SZ16);
  _Float16* vt  = (_Float16*)(ws + 3 * SZ16);
  _Float16* ctx = (_Float16*)(ws + 4 * SZ16);
  float*    hres = (float*)(ws + 5 * SZ16);
  _Float16* yln = (_Float16*)(ws + 7 * SZ16);
  _Float16* act = (_Float16*)(ws + 8 * SZ16);
  char* wbase   = ws + 12 * SZ16;                               // 96 MiB
  _Float16* WTq = (_Float16*)(wbase);
  _Float16* WTk = (_Float16*)(wbase + 1 * SZW);
  _Float16* WTv = (_Float16*)(wbase + 2 * SZW);
  _Float16* WTo = (_Float16*)(wbase + 3 * SZW);
  _Float16* WT1 = (_Float16*)(wbase + 4 * SZW);
  _Float16* WT2 = (_Float16*)(wbase + 8 * SZW);

  // one-time weight convert+transpose (f32 [K,N] -> f16 [N,K])
  wconv_kernel<<<dim3(HDIM / 32, HDIM / 32), 256, 0, stream>>>(Wq, WTq, HDIM, HDIM);
  wconv_kernel<<<dim3(HDIM / 32, HDIM / 32), 256, 0, stream>>>(Wk, WTk, HDIM, HDIM);
  wconv_kernel<<<dim3(HDIM / 32, HDIM / 32), 256, 0, stream>>>(Wv, WTv, HDIM, HDIM);
  wconv_kernel<<<dim3(HDIM / 32, HDIM / 32), 256, 0, stream>>>(Wo, WTo, HDIM, HDIM);
  wconv_kernel<<<dim3(FF / 32, HDIM / 32), 256, 0, stream>>>(W1, WT1, HDIM, FF);
  wconv_kernel<<<dim3(HDIM / 32, FF / 32), 256, 0, stream>>>(W2, WT2, FF, HDIM);

  ln_kernel<<<MTOT, 256, 0, stream>>>(hidden, g1, be1, xln);
  qkv_kernel<<<dim3(HDIM / 64, MTOT / 256, 3), 256, 0, stream>>>(
      xln, WTq, bq, WTk, bk, WTv, bv, qh, kh, vt);
  attn_kernel<<<dim3(SEQ / 64, 2 * NHEAD), 128, 0, stream>>>(qh, kh, vt, ctx);
  oproj_kernel<<<dim3(HDIM / 64, MTOT / 256), 256, 0, stream>>>(ctx, WTo, bo, hidden, hres);
  ln_kernel<<<MTOT, 256, 0, stream>>>(hres, g2, be2, yln);
  mlp1_kernel<<<dim3(FF / 64, MTOT / 256), 256, 0, stream>>>(yln, WT1, b1, act);
  mlp2_kernel<<<dim3(HDIM / 64, MTOT / 256), 256, 0, stream>>>(act, WT2, b2, hres, out);
}
